// AttentionSimpleRNN_90555090469052
// MI455X (gfx1250) — matrix operand order repaired and checked by reading, not compiled, not run on hardware
//
#include <hip/hip_runtime.h>
#include <hip/hip_bf16.h>
#include <cmath>

// ---------------------------------------------------------------------------
// AttentionSimpleRNN on gfx1250 (MI455X).
// B=64, T=512, D=256 everywhere. bf16 WMMA (v_wmma_f32_16x16x32_bf16) for all
// GEMM-shaped work (M batched into 16-row tiles), VALU for tanh/softmax.
// Weights pre-converted once into fragment-ready bf16 layout so B fragments
// are two global_load_b128 per lane with zero LDS staging / barriers.
// ---------------------------------------------------------------------------

#define BB   64
#define TT   512
#define DD   256

typedef __attribute__((ext_vector_type(16))) __bf16         v16bf;
typedef __attribute__((ext_vector_type(16))) unsigned short v16u;
typedef __attribute__((ext_vector_type(8)))  unsigned short v8u;
typedef __attribute__((ext_vector_type(4)))  unsigned short v4u;
typedef __attribute__((ext_vector_type(8)))  float          v8f;

__device__ __forceinline__ unsigned short f2bf(float f) {
  unsigned int u = __float_as_uint(f);
  u += 0x7fffu + ((u >> 16) & 1u);        // round-to-nearest-even
  return (unsigned short)(u >> 16);
}

__device__ __forceinline__ v4u f2bf4(float4 f) {
  v4u r;
  r[0] = f2bf(f.x); r[1] = f2bf(f.y); r[2] = f2bf(f.z); r[3] = f2bf(f.w);
  return r;
}

__device__ __forceinline__ v8f wmma_bf16(v16bf a, v16bf b, v8f c) {
  return __builtin_amdgcn_wmma_f32_16x16x32_bf16(
      /*neg_a=*/false, a, /*neg_b=*/false, b,
      /*c_mod=*/(short)0, c, /*reuse_a=*/false, /*reuse_b=*/false);
}

__device__ __forceinline__ v16bf merge16(v8u lo, v8u hi) {
  v16u t = __builtin_shufflevector(lo, hi, 0, 1, 2, 3, 4, 5, 6, 7,
                                   8, 9, 10, 11, 12, 13, 14, 15);
  return __builtin_bit_cast(v16bf, t);
}

// A fragment (16x32 bf16, MxK) from row-major LDS tile [16][DD], chunk at k0.
// lanes 0-15 -> row M=lane, K={k0..k0+7, k0+16..k0+23};
// lanes 16-31 -> row M=lane-16, K={k0+8..k0+15, k0+24..k0+31}.
// Both halves are 16B-aligned contiguous runs -> two ds_load_b128.
__device__ __forceinline__ v16bf load_a_frag(const unsigned short* A, int k0,
                                             int lane) {
  const unsigned short* p =
      A + (lane & 15) * DD + k0 + ((lane & 16) ? 8 : 0);
  return merge16(*(const v8u*)p, *(const v8u*)(p + 16));
}

// B fragment (32x16 bf16, KxN) straight from the prepped global bf16 weight
// image laid out as [kc][n][kk] (kk contiguous, 32 per n).
// lanes 0-15: column n0+lane, K={0..7, 16..23}; lanes 16-31: same column,
// K={8..15, 24..31}.
// One lane = two 16B runs -> two global_load_b128.
__device__ __forceinline__ v16bf load_b_global(const unsigned short* Bg,
                                               int kc, int n0, int lane) {
  const unsigned short* p =
      Bg + kc * (32 * DD) + (n0 + (lane & 15)) * 32 + ((lane & 16) ? 8 : 0);
  return merge16(*(const v8u*)p, *(const v8u*)(p + 16));
}

// ---------------------------------------------------------------------------
// Kernel 0: convert the five 256x256 f32 weight matrices into bf16 in
// fragment-ready layout dst[kc*8192 + n*32 + kk] = M[(kc*32+kk)*256 + n].
// One-time cost; result lives in L2 for all 512 steps.
// Grid: (32, 5), 256 threads.
// ---------------------------------------------------------------------------
__global__ __launch_bounds__(256) void prep_weights_kernel(
    const float* __restrict__ W, const float* __restrict__ U,
    const float* __restrict__ Cm, const float* __restrict__ Ua,
    const float* __restrict__ Wa, unsigned short* __restrict__ dst) {
  const float* srcs[5] = {W, U, Cm, Ua, Wa};
  const float* src = srcs[blockIdx.y];
  unsigned short* d = dst + (size_t)blockIdx.y * DD * DD;
  for (int r = 0; r < 8; ++r) {
    int i  = blockIdx.x * 2048 + r * 256 + threadIdx.x;
    int kk = i & 31, n = (i >> 5) & 255, kc = i >> 13;
    d[i] = f2bf(src[(size_t)(kc * 32 + kk) * DD + n]);
  }
}

// ---------------------------------------------------------------------------
// Kernel 1: xWa = x @ Wa.  x viewed as (B*T) x 256 row-major.
// Grid: 2048 WGs (one 16-row M tile each), 256 threads = 8 waves,
// wave w owns columns [32w, 32w+32). Barrier-free k-loop.
// ---------------------------------------------------------------------------
__global__ __launch_bounds__(256) void xwa_kernel(
    const float* __restrict__ x, const unsigned short* __restrict__ Wab,
    float* __restrict__ xWa) {
  __shared__ unsigned short As[16 * DD];   // [m][k] bf16

  const int m0   = blockIdx.x * 16;
  const int tid  = threadIdx.x;
  const int w    = tid >> 5;
  const int lane = tid & 31;
  const int n0   = w * 32;

  // Stage A: 1024 float4 -> packed v4u LDS stores.
  for (int j = tid; j < (16 * DD) / 4; j += 256) {
    int m = j >> 6, k4 = (j & 63) * 4;
    float4 xv = *(const float4*)&x[(size_t)(m0 + m) * DD + k4];
    ((v4u*)As)[j] = f2bf4(xv);
  }
  __syncthreads();

  v8f acc0 = {}, acc1 = {};
#pragma unroll
  for (int kc = 0; kc < 8; ++kc) {
    v16bf a = load_a_frag(As, kc * 32, lane);
    acc0 = wmma_bf16(a, load_b_global(Wab, kc, n0, lane), acc0);
    acc1 = wmma_bf16(a, load_b_global(Wab, kc, n0 + 16, lane), acc1);
  }

  const int nl = lane & 15;
  const int mb = (lane & 16) ? 8 : 0;
#pragma unroll
  for (int r = 0; r < 8; ++r) {
    int m = mb + r;
    xWa[(size_t)(m0 + m) * DD + n0 + nl]      = acc0[r];
    xWa[(size_t)(m0 + m) * DD + n0 + 16 + nl] = acc1[r];
  }
}

// ---------------------------------------------------------------------------
// Kernel 2 (per step): e[b,tau] = sum_a tanh(xWa[b,tau,a] + hUa[b,a]) * va[a]
// Grid: (8 tau-chunks, 64 batches), 256 threads. Wave handles 8 taus; lane l
// covers a = l + 32s (coalesced), then 5-step shfl_xor reduce.
// ---------------------------------------------------------------------------
__global__ __launch_bounds__(256) void scores_kernel(
    const float* __restrict__ xWa, const float* __restrict__ hUa,
    const float* __restrict__ va, float* __restrict__ e) {
  __shared__ float sh_h[DD];
  __shared__ float sh_v[DD];
  const int b   = blockIdx.y;
  const int tc  = blockIdx.x;      // 64 taus per block
  const int tid = threadIdx.x;
  sh_h[tid] = hUa[b * DD + tid];
  sh_v[tid] = va[tid];
  __syncthreads();

  const int w = tid >> 5, l = tid & 31;
  const float* base = xWa + (size_t)b * TT * DD;
#pragma unroll 1
  for (int t8 = 0; t8 < 8; ++t8) {
    int tau = tc * 64 + w * 8 + t8;
    const float* row = base + (size_t)tau * DD;
    float p = 0.f;
#pragma unroll
    for (int s = 0; s < 8; ++s) {
      int a = l + 32 * s;
      p = fmaf(tanhf(row[a] + sh_h[a]), sh_v[a], p);
    }
#pragma unroll
    for (int off = 16; off > 0; off >>= 1) p += __shfl_xor(p, off, 32);
    if (l == 0) e[b * TT + tau] = p;
  }
}

// ---------------------------------------------------------------------------
// Kernel 3 (per step): alpha = softmax(e[b,:]); ctx[b,i] = sum_tau alpha*x.
// Grid: 64 WGs (one per batch), 256 threads.
// ---------------------------------------------------------------------------
__global__ __launch_bounds__(256) void softmax_ctx_kernel(
    const float* __restrict__ e, const float* __restrict__ x,
    float* __restrict__ ctx) {
  __shared__ float sa[TT];
  __shared__ float red[256];
  const int b = blockIdx.x, tid = threadIdx.x;

  float e0 = e[b * TT + tid];
  float e1 = e[b * TT + 256 + tid];
  red[tid] = fmaxf(e0, e1);
  __syncthreads();
  for (int s = 128; s > 0; s >>= 1) {
    if (tid < s) red[tid] = fmaxf(red[tid], red[tid + s]);
    __syncthreads();
  }
  const float mx = red[0];
  __syncthreads();

  float x0 = __expf(e0 - mx), x1 = __expf(e1 - mx);
  red[tid] = x0 + x1;
  __syncthreads();
  for (int s = 128; s > 0; s >>= 1) {
    if (tid < s) red[tid] += red[tid + s];
    __syncthreads();
  }
  const float inv = 1.0f / red[0];
  sa[tid]       = x0 * inv;
  sa[256 + tid] = x1 * inv;
  __syncthreads();

  // ctx: thread owns channel i = tid; coalesced across threads per tau.
  const float* xb = x + (size_t)b * TT * DD;
  float acc = 0.f;
#pragma unroll 4
  for (int tau = 0; tau < TT; ++tau)
    acc = fmaf(sa[tau], xb[(size_t)tau * DD + tid], acc);
  ctx[b * DD + tid] = acc;
}

// ---------------------------------------------------------------------------
// Kernel 4 (per step): h_new = tanh(Xt@W + H@U + Ctx@C + bias) for a 16-batch
// tile (3 accumulated WMMA GEMMs, barrier-free k-loop), write h history to
// out[:,t,:], then hUa_next = h_new @ Ua (4th WMMA GEMM) for the next step.
// Grid: 4 WGs (batch tiles), 256 threads = 8 waves, wave w owns cols
// [32w, 32w+32).
// ---------------------------------------------------------------------------
__global__ __launch_bounds__(256) void update_kernel(
    const float* __restrict__ x, const unsigned short* __restrict__ Wb,
    const unsigned short* __restrict__ Ub, const unsigned short* __restrict__ Cb,
    const float* __restrict__ bias, const unsigned short* __restrict__ Uab,
    const float* __restrict__ ctx, float* __restrict__ h,
    float* __restrict__ hUa, float* __restrict__ out, int t) {
  __shared__ unsigned short Xa[16 * DD];  // Xt bf16; reused later for h_new
  __shared__ unsigned short Ha[16 * DD];  // h    bf16
  __shared__ unsigned short Ca[16 * DD];  // ctx  bf16

  const int b0   = blockIdx.x * 16;
  const int tid  = threadIdx.x;
  const int w    = tid >> 5;
  const int lane = tid & 31;
  const int n0   = w * 32;

  // Stage A operands (snapshot h before we overwrite it below).
  for (int j = tid; j < (16 * DD) / 4; j += 256) {
    int m = j >> 6, k4 = (j & 63) * 4;
    float4 xv = *(const float4*)&x[((size_t)(b0 + m) * TT + t) * DD + k4];
    float4 hv = *(const float4*)&h[(b0 + m) * DD + k4];
    float4 cv = *(const float4*)&ctx[(b0 + m) * DD + k4];
    ((v4u*)Xa)[j] = f2bf4(xv);
    ((v4u*)Ha)[j] = f2bf4(hv);
    ((v4u*)Ca)[j] = f2bf4(cv);
  }
  __syncthreads();

  v8f acc0 = {}, acc1 = {};
#pragma unroll
  for (int kc = 0; kc < 8; ++kc) {
    v16bf ax = load_a_frag(Xa, kc * 32, lane);
    v16bf ah = load_a_frag(Ha, kc * 32, lane);
    v16bf ac = load_a_frag(Ca, kc * 32, lane);
    acc0 = wmma_bf16(ax, load_b_global(Wb, kc, n0, lane), acc0);
    acc1 = wmma_bf16(ax, load_b_global(Wb, kc, n0 + 16, lane), acc1);
    acc0 = wmma_bf16(ah, load_b_global(Ub, kc, n0, lane), acc0);
    acc1 = wmma_bf16(ah, load_b_global(Ub, kc, n0 + 16, lane), acc1);
    acc0 = wmma_bf16(ac, load_b_global(Cb, kc, n0, lane), acc0);
    acc1 = wmma_bf16(ac, load_b_global(Cb, kc, n0 + 16, lane), acc1);
  }
  __syncthreads();  // all WMMA reads of Xa done before Xa is reused

  const int   nl  = lane & 15;
  const int   mb  = (lane & 16) ? 8 : 0;
  const float bv0 = bias[n0 + nl];
  const float bv1 = bias[n0 + 16 + nl];
#pragma unroll
  for (int r = 0; r < 8; ++r) {
    int   m  = mb + r;
    float h0 = tanhf(acc0[r] + bv0);
    float h1 = tanhf(acc1[r] + bv1);
    size_t orow = ((size_t)(b0 + m) * TT + t) * DD;
    out[orow + n0 + nl]      = h0;
    out[orow + n0 + 16 + nl] = h1;
    h[(b0 + m) * DD + n0 + nl]      = h0;
    h[(b0 + m) * DD + n0 + 16 + nl] = h1;
    Xa[m * DD + n0 + nl]      = f2bf(h0);   // h_new bf16 for the Ua GEMM
    Xa[m * DD + n0 + 16 + nl] = f2bf(h1);
  }
  __syncthreads();

  // hUa_next = h_new @ Ua
  v8f acc2 = {}, acc3 = {};
#pragma unroll
  for (int kc = 0; kc < 8; ++kc) {
    v16bf a = load_a_frag(Xa, kc * 32, lane);
    acc2 = wmma_bf16(a, load_b_global(Uab, kc, n0, lane), acc2);
    acc3 = wmma_bf16(a, load_b_global(Uab, kc, n0 + 16, lane), acc3);
  }
#pragma unroll
  for (int r = 0; r < 8; ++r) {
    int m = mb + r;
    hUa[(b0 + m) * DD + n0 + nl]      = acc2[r];
    hUa[(b0 + m) * DD + n0 + 16 + nl] = acc3[r];
  }
}

// ---------------------------------------------------------------------------
// Host launcher. Inputs: x, W, U, C, b, Wa, Ua, va (all float32).
// ws layout: xWa[B*T*D] f32 | e[B*T] f32 | hUa[B*D] f32 | ctx[B*D] f32 |
//            h[B*D] f32 | Wbf[5][256*256] bf16 (order: W, U, C, Ua, Wa)
// ---------------------------------------------------------------------------
extern "C" void kernel_launch(void* const* d_in, const int* in_sizes, int n_in,
                              void* d_out, int out_size, void* d_ws,
                              size_t ws_size, hipStream_t stream) {
  (void)in_sizes; (void)n_in; (void)out_size; (void)ws_size;
  const float* x    = (const float*)d_in[0];
  const float* W    = (const float*)d_in[1];
  const float* U    = (const float*)d_in[2];
  const float* Cm   = (const float*)d_in[3];
  const float* bias = (const float*)d_in[4];
  const float* Wa   = (const float*)d_in[5];
  const float* Ua   = (const float*)d_in[6];
  const float* va   = (const float*)d_in[7];
  float*       out  = (float*)d_out;

  float* ws   = (float*)d_ws;
  float* xWa  = ws;                                  // B*T*D
  float* e    = xWa + (size_t)BB * TT * DD;          // B*T
  float* hUa  = e + (size_t)BB * TT;                 // B*D
  float* ctxb = hUa + (size_t)BB * DD;               // B*D
  float* h    = ctxb + (size_t)BB * DD;              // B*D
  unsigned short* wbf = (unsigned short*)(h + (size_t)BB * DD);
  unsigned short* Wb  = wbf + 0 * (size_t)DD * DD;
  unsigned short* Ub  = wbf + 1 * (size_t)DD * DD;
  unsigned short* Cb  = wbf + 2 * (size_t)DD * DD;
  unsigned short* Uab = wbf + 3 * (size_t)DD * DD;
  unsigned short* Wab = wbf + 4 * (size_t)DD * DD;

  // h0 = 0 -> h, hUa (=0@Ua), ctx scratch all zeroed (contiguous region).
  hipMemsetAsync(hUa, 0, (size_t)3 * BB * DD * sizeof(float), stream);

  // One-time weight conversion into fragment-ready bf16 layout.
  prep_weights_kernel<<<dim3(32, 5), 256, 0, stream>>>(W, U, Cm, Ua, Wa, wbf);

  // Precompute attention key projection of the whole sequence.
  xwa_kernel<<<(BB * TT) / 16, 256, 0, stream>>>(x, Wab, xWa);

  for (int t = 0; t < TT; ++t) {
    scores_kernel<<<dim3(8, BB), 256, 0, stream>>>(xWa, hUa, va, e);
    softmax_ctx_kernel<<<BB, 256, 0, stream>>>(e, x, ctxb);
    update_kernel<<<BB / 16, 256, 0, stream>>>(x, Wb, Ub, Cb, bias, Uab, ctxb,
                                               h, hUa, out, t);
  }
}
